// RGLRU_60490319396914
// MI455X (gfx1250) — matrix operand order repaired and checked by reading, not compiled, not run on hardware
//
#include <hip/hip_runtime.h>
#include <cstdint>

// ---------------- problem constants ----------------
#define B_ 4
#define L_ 4096
#define D_ 1024
#define M_ (B_ * L_)   // 16384 rows
#define K_ D_          // 1024 reduction

// ---------------- gemm tiling ----------------
#define BM 128
#define BN 64
#define BK 32
#define BKP 40             // padded LDS row stride (80B = 20 dwords -> conflict-free)
#define NSTAGE (K_ / BK)   // 32

// ---------------- scan tiling ----------------
#define SEG 64
#define SEGLEN (L_ / SEG)  // 64

typedef unsigned int u32;
typedef __attribute__((ext_vector_type(16))) __bf16 v16bf;
typedef __attribute__((ext_vector_type(8)))  __bf16 v8bf;
typedef __attribute__((ext_vector_type(8)))  float  v8f;
typedef __attribute__((ext_vector_type(4)))  u32    u32x4;
typedef __attribute__((ext_vector_type(8)))  int    i32x8;
typedef __attribute__((ext_vector_type(4)))  int    i32x4;

// ---------------- TDM availability ----------------
#if defined(__has_builtin)
#if __has_builtin(__builtin_amdgcn_tensor_load_to_lds) && __has_builtin(__builtin_amdgcn_s_wait_tensorcnt)
#define USE_TDM 1
#endif
#endif
#ifndef USE_TDM
#define USE_TDM 0
#endif

#if USE_TDM
__device__ __forceinline__ void tdm_call(u32x4 g0, i32x8 g1, i32x4 g2, i32x4 g3) {
#if defined(__clang_major__) && __clang_major__ >= 23
  i32x8 g4 = {0, 0, 0, 0, 0, 0, 0, 0};
  __builtin_amdgcn_tensor_load_to_lds(g0, g1, g2, g3, g4, 0);
#else
  __builtin_amdgcn_tensor_load_to_lds(g0, g1, g2, g3, 0);
#endif
}

// Build a D# (ISA 8.3-8.6) for a bf16 (data_size=1 -> 2B) tile load into LDS.
// tile dims (td0,td1,td2) and tensor dims (xd0,xd1,xd2) in elements,
// strides (s0 = dim0 stride, s1 = dim1 stride) in elements.
// LDS destination is padded: after every 64B (one 32-elem row: pad_interval=3)
// insert 16B (4 dwords: pad_amount=3) -> 80B effective row stride (BKP=40).
__device__ __forceinline__ void tdm_load_bf16(u32 lds_addr, uint64_t gaddr,
                                              u32 td0, u32 td1, u32 td2,
                                              u32 xd0, u32 xd1, u32 xd2,
                                              uint64_t s0, uint64_t s1) {
  u32x4 g0;
  g0.x = 1u;                                  // count=1, user mode, no gather
  g0.y = lds_addr;                            // lds_addr [63:32]
  g0.z = (u32)gaddr;                          // global_addr[31:0]
  g0.w = ((u32)(gaddr >> 32) & 0x01FFFFFFu) | (2u << 30);  // addr[56:32] | type=2

  i32x8 g1;
  g1[0] = (int)((1u << 16) |                  // data_size = 1 (2 bytes)
                (1u << 20) |                  // pad_enable
                (3u << 22) |                  // pad_interval: 16 dwords (64B)
                (3u << 25));                  // pad_amount: 4 dwords (16B)
  g1[1] = (int)((xd0 & 0xFFFFu) << 16);       // tensor_dim0[15:0] at bits 63:48
  g1[2] = (int)(((xd0 >> 16) & 0xFFFFu) | ((xd1 & 0xFFFFu) << 16));
  g1[3] = (int)(((xd1 >> 16) & 0xFFFFu) | ((td0 & 0xFFFFu) << 16));
  g1[4] = (int)((td1 & 0xFFFFu) | ((td2 & 0xFFFFu) << 16));
  g1[5] = (int)(u32)(s0 & 0xFFFFFFFFull);
  g1[6] = (int)((u32)((s0 >> 32) & 0xFFFFull) | ((u32)(s1 & 0xFFFFull) << 16));
  g1[7] = (int)(u32)((s1 >> 16) & 0xFFFFFFFFull);

  i32x4 g2 = {(int)xd2, 0, 0, 0};             // tensor_dim2; no dim3/iterate
  i32x4 g3 = {0, 0, 0, 0};
  tdm_call(g0, g1, g2, g3);
}
#endif  // USE_TDM

// ---------------- helpers ----------------
__device__ __forceinline__ float sigmoidf_(float v) {
  return 1.0f / (1.0f + __expf(-v));
}

__device__ __forceinline__ v8f wmma_bf16(v16bf a, v16bf b, v8f c) {
  return __builtin_amdgcn_wmma_f32_16x16x32_bf16(false, a, false, b, (short)0, c,
                                                 false, false);
}

// A-fragment (16x32 bf16): lane half h needs K = [8h..8h+7] and [16+8h..16+8h+7]
__device__ __forceinline__ v16bf load_afrag(const __bf16* row, int half) {
  v8bf lo = *(const v8bf*)(row + 8 * half);
  v8bf hi = *(const v8bf*)(row + 16 + 8 * half);
  return __builtin_shufflevector(lo, hi, 0, 1, 2, 3, 4, 5, 6, 7, 8, 9, 10, 11,
                                 12, 13, 14, 15);
}

// B-fragment (32x16 bf16): lane = column n, half h needs K = [8h..8h+7] and [16+8h..16+8h+7]
// (two 16B loads: padded rows are only 16B aligned)
__device__ __forceinline__ v16bf load_bfrag(const __bf16* row, int half) {
  v8bf lo = *(const v8bf*)(row + 8 * half);
  v8bf hi = *(const v8bf*)(row + 16 + 8 * half);
  return __builtin_shufflevector(lo, hi, 0, 1, 2, 3, 4, 5, 6, 7, 8, 9, 10, 11,
                                 12, 13, 14, 15);
}

// ---------------- kernel 1: f32 -> (hi,lo) bf16 split ----------------
__global__ void split_bf16_kernel(const float* __restrict__ src,
                                  __bf16* __restrict__ hi,
                                  __bf16* __restrict__ lo, int n) {
  int t = blockIdx.x * blockDim.x + threadIdx.x;
  if (t < n) {
    float v = src[t];
    __bf16 h = (__bf16)v;
    hi[t] = h;
    lo[t] = (__bf16)(v - (float)h);
  }
}

// ---------------- kernel 2: fused dual GEMM + gates ----------------
// p_r = x @ Wa^T, p_i = x @ Wx^T   (W rows are K-contiguous == B columns)
// a_t = exp(8 * sigmoid(p_r+ba) * log(sigmoid(lmbd)))
// b_t = sqrt(1-a_t^2) * sigmoid(p_i+bx) * x
__global__ __launch_bounds__(256) void gemm_gates_kernel(
    const __bf16* __restrict__ xh, const __bf16* __restrict__ xl,
    const __bf16* __restrict__ wh, const __bf16* __restrict__ wl,
    const float* __restrict__ x, const float* __restrict__ ba,
    const float* __restrict__ bx, const float* __restrict__ lmbd,
    float* __restrict__ at_out, float* __restrict__ bt_out) {
  __shared__ __attribute__((aligned(32))) __bf16 sXh[2][BM][BKP];
  __shared__ __attribute__((aligned(32))) __bf16 sXl[2][BM][BKP];
  __shared__ __attribute__((aligned(32))) __bf16 sWh[2][2][BN][BKP];
  __shared__ __attribute__((aligned(32))) __bf16 sWl[2][2][BN][BKP];

  const int tid = threadIdx.x;
  const int w = tid >> 5;       // wave 0..7
  const int lane = tid & 31;
  const int half = lane >> 4;
  const int l16 = lane & 15;
  const int wm = w >> 1;        // 0..3 -> 32-row slice
  const int wn = w & 1;         // 0..1 -> 32-col slice
  const int m0 = blockIdx.x * BM;
  const int n0 = blockIdx.y * BN;
  // force a *scalar* branch for TDM issuance (TDM ignores EXEC, so an
  // exec-masked fallthrough would duplicate the DMA 8x)
  const bool issuer = (__builtin_amdgcn_readfirstlane(w) == 0);

  v8f acc_r[2][2], acc_i[2][2];
#pragma unroll
  for (int mt = 0; mt < 2; ++mt)
#pragma unroll
    for (int nt = 0; nt < 2; ++nt) {
      acc_r[mt][nt] = 0;
      acc_i[mt][nt] = 0;
    }

#if USE_TDM
  auto issue_stage = [&](int st) {
    const int buf = st & 1;
    const uint64_t k0 = (uint64_t)st * BK;
    // X hi/lo tiles: 2D tile 32(K) x 128(M), global row stride K_
    tdm_load_bf16((u32)(uintptr_t)&sXh[buf][0][0],
                  (uint64_t)(uintptr_t)xh + ((uint64_t)m0 * K_ + k0) * 2,
                  BK, BM, 0, K_, M_, 0, (uint64_t)K_, 0);
    tdm_load_bf16((u32)(uintptr_t)&sXl[buf][0][0],
                  (uint64_t)(uintptr_t)xl + ((uint64_t)m0 * K_ + k0) * 2,
                  BK, BM, 0, K_, M_, 0, (uint64_t)K_, 0);
    // W hi/lo tiles: 3D tile 32(K) x 64(N) x 2(Wa,Wx); plane stride D_*K_
    tdm_load_bf16((u32)(uintptr_t)&sWh[buf][0][0][0],
                  (uint64_t)(uintptr_t)wh + ((uint64_t)n0 * K_ + k0) * 2,
                  BK, BN, 2, K_, 2 * D_, 2, (uint64_t)K_, (uint64_t)D_ * K_);
    tdm_load_bf16((u32)(uintptr_t)&sWl[buf][0][0][0],
                  (uint64_t)(uintptr_t)wl + ((uint64_t)n0 * K_ + k0) * 2,
                  BK, BN, 2, K_, 2 * D_, 2, (uint64_t)K_, (uint64_t)D_ * K_);
  };
  if (issuer) issue_stage(0);
#endif

  for (int s = 0; s < NSTAGE; ++s) {
    const int buf = s & 1;
#if USE_TDM
    if (issuer) {
      if (s + 1 < NSTAGE) {
        issue_stage(s + 1);
        __builtin_amdgcn_s_wait_tensorcnt(4);  // stage s complete (4 newest pending)
      } else {
        __builtin_amdgcn_s_wait_tensorcnt(0);
      }
    }
    __syncthreads();
#else
    __syncthreads();
    const int k0 = s * BK;
    for (int idx = tid; idx < BM * BK; idx += 256) {
      int r = idx >> 5, c = idx & (BK - 1);
      size_t g = (size_t)(m0 + r) * K_ + k0 + c;
      sXh[buf][r][c] = xh[g];
      sXl[buf][r][c] = xl[g];
    }
    for (int idx = tid; idx < 2 * BN * BK; idx += 256) {
      int gs = idx >> 11, rem = idx & 2047;
      int r = rem >> 5, c = rem & (BK - 1);
      size_t g = (size_t)(gs * D_ + n0 + r) * K_ + k0 + c;
      sWh[buf][gs][r][c] = wh[g];
      sWl[buf][gs][r][c] = wl[g];
    }
    __syncthreads();
#endif

    v16bf ah[2], al[2], bhf[2][2], blf[2][2];
#pragma unroll
    for (int mt = 0; mt < 2; ++mt) {
      int mr = wm * 32 + mt * 16 + l16;
      ah[mt] = load_afrag(&sXh[buf][mr][0], half);
      al[mt] = load_afrag(&sXl[buf][mr][0], half);
    }
#pragma unroll
    for (int g = 0; g < 2; ++g)
#pragma unroll
      for (int nt = 0; nt < 2; ++nt) {
        int nr = wn * 32 + nt * 16 + l16;
        bhf[g][nt] = load_bfrag(&sWh[buf][g][nr][0], half);
        blf[g][nt] = load_bfrag(&sWl[buf][g][nr][0], half);
      }
#pragma unroll
    for (int mt = 0; mt < 2; ++mt)
#pragma unroll
      for (int nt = 0; nt < 2; ++nt) {
        acc_r[mt][nt] = wmma_bf16(ah[mt], bhf[0][nt], acc_r[mt][nt]);
        acc_r[mt][nt] = wmma_bf16(ah[mt], blf[0][nt], acc_r[mt][nt]);
        acc_r[mt][nt] = wmma_bf16(al[mt], bhf[0][nt], acc_r[mt][nt]);
        acc_i[mt][nt] = wmma_bf16(ah[mt], bhf[1][nt], acc_i[mt][nt]);
        acc_i[mt][nt] = wmma_bf16(ah[mt], blf[1][nt], acc_i[mt][nt]);
        acc_i[mt][nt] = wmma_bf16(al[mt], bhf[1][nt], acc_i[mt][nt]);
      }
    __syncthreads();
  }

  // epilogue: gates + a_t/b_t (C/D layout: VGPR v -> M = v + 8*half, N = lane&15)
#pragma unroll
  for (int nt = 0; nt < 2; ++nt) {
    int n = n0 + wn * 32 + nt * 16 + l16;
    float lam = lmbd[n];
    float la = __logf(sigmoidf_(lam));  // log(sigmoid(lmbd)) < 0
    float bav = ba[n], bxv = bx[n];
#pragma unroll
    for (int mt = 0; mt < 2; ++mt)
#pragma unroll
      for (int v = 0; v < 8; ++v) {
        int m = m0 + wm * 32 + mt * 16 + 8 * half + v;
        float r = sigmoidf_(acc_r[mt][nt][v] + bav);
        float ig = sigmoidf_(acc_i[mt][nt][v] + bxv);
        float a = __expf(8.0f * r * la);
        float xv = x[(size_t)m * D_ + n];
        float bt = sqrtf(fmaxf(1.0f - a * a, 0.0f)) * ig * xv;
        at_out[(size_t)m * D_ + n] = a;
        bt_out[(size_t)m * D_ + n] = bt;
      }
  }
}

// ---------------- kernel 3a: per-segment reduce (A = prod a, H = local scan end) ----
__global__ void seg_reduce_kernel(const float* __restrict__ at,
                                  const float* __restrict__ bt,
                                  float* __restrict__ Aseg,
                                  float* __restrict__ Hseg) {
  int t = blockIdx.x * blockDim.x + threadIdx.x;  // B*SEG*D threads
  int d = t & (D_ - 1);
  int bs = t >> 10;
  int seg = bs & (SEG - 1);
  int b = bs >> 6;
  size_t base = ((size_t)b * L_ + (size_t)seg * SEGLEN) * D_ + d;
  float A = 1.0f, h = 0.0f;
  for (int i = 0; i < SEGLEN; ++i) {
    float a = at[base + (size_t)i * D_];
    float bb = bt[base + (size_t)i * D_];
    h = a * h + bb;
    A *= a;
  }
  Aseg[t] = A;
  Hseg[t] = h;
}

// ---------------- kernel 3b: scan across segments -> carry-in per segment ----------
__global__ void seg_scan_kernel(const float* __restrict__ Aseg,
                                const float* __restrict__ Hseg,
                                float* __restrict__ carry) {
  int t = blockIdx.x * blockDim.x + threadIdx.x;  // B*D threads
  int d = t & (D_ - 1);
  int b = t >> 10;
  float h = 0.0f;
  for (int s = 0; s < SEG; ++s) {
    size_t idx = ((size_t)(b * SEG + s)) * D_ + d;
    carry[idx] = h;
    h = Aseg[idx] * h + Hseg[idx];
  }
}

// ---------------- kernel 3c: apply recurrence with carry, write y ------------------
__global__ void seg_apply_kernel(const float* __restrict__ at,
                                 const float* __restrict__ bt,
                                 const float* __restrict__ carry,
                                 float* __restrict__ y) {
  int t = blockIdx.x * blockDim.x + threadIdx.x;  // B*SEG*D threads
  int d = t & (D_ - 1);
  int bs = t >> 10;
  int seg = bs & (SEG - 1);
  int b = bs >> 6;
  size_t base = ((size_t)b * L_ + (size_t)seg * SEGLEN) * D_ + d;
  float h = carry[t];
  for (int i = 0; i < SEGLEN; ++i) {
    float a = at[base + (size_t)i * D_];
    float bb = bt[base + (size_t)i * D_];
    h = a * h + bb;
    y[base + (size_t)i * D_] = h;
  }
}

// ---------------- launch ----------------
extern "C" void kernel_launch(void* const* d_in, const int* in_sizes, int n_in,
                              void* d_out, int out_size, void* d_ws,
                              size_t ws_size, hipStream_t stream) {
  (void)in_sizes; (void)n_in; (void)out_size; (void)ws_size;
  const float* x    = (const float*)d_in[0];
  const float* Wa   = (const float*)d_in[1];
  const float* Wx   = (const float*)d_in[2];
  const float* ba   = (const float*)d_in[3];
  const float* bx   = (const float*)d_in[4];
  const float* lmbd = (const float*)d_in[5];

  char* ws = (char*)d_ws;
  size_t off = 0;
  __bf16* xh = (__bf16*)(ws + off); off += (size_t)M_ * K_ * 2;
  __bf16* xl = (__bf16*)(ws + off); off += (size_t)M_ * K_ * 2;
  __bf16* wh = (__bf16*)(ws + off); off += (size_t)2 * D_ * K_ * 2;
  __bf16* wl = (__bf16*)(ws + off); off += (size_t)2 * D_ * K_ * 2;
  float* at   = (float*)(ws + off); off += (size_t)M_ * D_ * 4;
  float* bt   = (float*)(ws + off); off += (size_t)M_ * D_ * 4;
  float* Aseg = (float*)(ws + off); off += (size_t)B_ * SEG * D_ * 4;
  float* Hseg = (float*)(ws + off); off += (size_t)B_ * SEG * D_ * 4;
  float* carry = (float*)(ws + off);

  // 1) split f32 -> bf16 hi/lo (x, Wa, Wx; Wa/Wx concatenated along rows)
  split_bf16_kernel<<<(M_ * K_ + 255) / 256, 256, 0, stream>>>(x, xh, xl, M_ * K_);
  split_bf16_kernel<<<(D_ * K_ + 255) / 256, 256, 0, stream>>>(Wa, wh, wl, D_ * K_);
  split_bf16_kernel<<<(D_ * K_ + 255) / 256, 256, 0, stream>>>(
      Wx, wh + (size_t)D_ * K_, wl + (size_t)D_ * K_, D_ * K_);

  // 2) fused dual-GEMM (WMMA bf16 split precision) + gate epilogue
  gemm_gates_kernel<<<dim3(M_ / BM, D_ / BN), 256, 0, stream>>>(
      xh, xl, wh, wl, x, ba, bx, lmbd, at, bt);

  // 3) segmented linear recurrence
  seg_reduce_kernel<<<(B_ * SEG * D_) / 256, 256, 0, stream>>>(at, bt, Aseg, Hseg);
  seg_scan_kernel<<<(B_ * D_) / 256, 256, 0, stream>>>(Aseg, Hseg, carry);
  seg_apply_kernel<<<(B_ * SEG * D_) / 256, 256, 0, stream>>>(at, bt, carry,
                                                              (float*)d_out);
}
